// Tacotron_82944408420359
// MI455X (gfx1250) — compile-verified, matrix operand order checked by reading, not run on hardware
//
#include <hip/hip_runtime.h>
#include <hip/hip_bf16.h>

typedef _Float16 v8h  __attribute__((ext_vector_type(8)));
typedef _Float16 v16h __attribute__((ext_vector_type(16)));
typedef float    v8f  __attribute__((ext_vector_type(8)));

#define B_   128
#define TENC 1024
#define DENC 256
#define MEL_ 80
#define NOUT 400
#define TDEC 250

__device__ __forceinline__ float fast_tanh(float x){
  // tanh(x) = 1 - 2/(exp(2x)+1), exp via v_exp_f32
  float e = __builtin_amdgcn_exp2f(x * 2.8853900817779268f);
  return 1.0f - 2.0f / (e + 1.0f);
}
__device__ __forceinline__ float fast_sigmoid(float x){
  return 1.0f / (1.0f + __builtin_amdgcn_exp2f(-1.4426950408889634f * x));
}

// One 16x16 output tile of  C += A(16xK, row-major f16) @ Bpk(packed)
// A fragment per ISA 7.12.2: lane<16 -> row=lane, K-halves {0..7,16..23};
// lane>=16 -> row=lane-16, K-halves {8..15,24..31}; per k-tile of 32.
// Bpk is pre-packed so each lane loads 16 contiguous halfs.
__device__ __forceinline__ v8f wmma_tile(const _Float16* __restrict__ A, int lda,
                                         const _Float16* __restrict__ Bpk,
                                         int KT, int nt, int lane){
  v8f acc = {};
  const int m  = lane & 15;
  const int kh = lane >> 4;
  const _Float16* arow = A + m * lda + kh * 8;
  for (int kt = 0; kt < KT; ++kt){
    v8h alo = *(const v8h*)(arow + kt*32);
    v8h ahi = *(const v8h*)(arow + kt*32 + 16);
    v16h a = __builtin_shufflevector(alo, ahi, 0,1,2,3,4,5,6,7,8,9,10,11,12,13,14,15);
    const _Float16* bp = Bpk + (((size_t)nt * KT + kt) * 32 + lane) * 16;
    v8h blo = *(const v8h*)(bp);
    v8h bhi = *(const v8h*)(bp + 8);
    v16h b = __builtin_shufflevector(blo, bhi, 0,1,2,3,4,5,6,7,8,9,10,11,12,13,14,15);
    acc = __builtin_amdgcn_wmma_f32_16x16x32_f16(false, a, false, b, (short)0, acc,
                                                 false, false);
  }
  return acc;
}

// ---------------- setup kernels ----------------

__global__ void k_init(const float* inp_att, const float* style_token,
                       float* ss, float* h_att, float* h1, float* h2, float* ctx,
                       float* last_out, float* wp){
  int i = blockIdx.x*256 + threadIdx.x;          // covers 128*256
  if (i < B_*DENC){
    int b = i / DENC, d = i % DENC;
    float s = 0.f;
    #pragma unroll
    for (int k = 0; k < 10; ++k) s += inp_att[b*10+k] * style_token[k*DENC+d];
    ss[i] = s; h_att[i] = 0.f; h1[i] = 0.f; h2[i] = 0.f; ctx[i] = 0.f;
  }
  if (i < B_*MEL_) last_out[i] = 0.f;
  if (i < TDEC)    wp[i] = 0.f;
}

__global__ void k_cvt(const float* x, _Float16* y, size_t n){
  size_t i = (size_t)blockIdx.x*256 + threadIdx.x;
  if (i < n) y[i] = (_Float16)x[i];
}

// pack row-major W[K][N] (zero-padded K) into WMMA B-fragment layout
__global__ void k_pack(const float* W, _Float16* out, int K, int N, int KT){
  int idx = blockIdx.x*256 + threadIdx.x;
  int tot = KT*32*N;
  if (idx >= tot) return;
  int j    = idx & 15;
  int lane = (idx >> 4) & 31;
  int kt   = (idx >> 9) % KT;
  int nt   = idx / (KT*512);
  int kh   = lane >> 4;
  int k = kt*32 + ((j < 8) ? (kh*8 + j) : (16 + kh*8 + (j-8)));
  int n = nt*16 + (lane & 15);
  out[idx] = (k < K) ? (_Float16)W[(size_t)k*N + n] : (_Float16)0.0f;
}

// keys = enc @ Wk + bk   (131072 x 256 x 256), f16 in, f16 out
__global__ void __launch_bounds__(256) k_keys(const _Float16* __restrict__ encH,
                                              const _Float16* __restrict__ Wkpk,
                                              const float* __restrict__ bk,
                                              _Float16* __restrict__ keysH){
  const int lane = threadIdx.x & 31, wave = threadIdx.x >> 5;
  const int tile = blockIdx.x*8 + wave;          // 131072 tiles
  const int mt = tile >> 4, nt = tile & 15;
  const _Float16* A = encH + (size_t)mt*16*256;
  v8f acc = wmma_tile(A, 256, Wkpk, 8, nt, lane);
  #pragma unroll
  for (int r = 0; r < 8; ++r){
    int m = (lane < 16) ? r : r+8, n = nt*16 + (lane & 15);
    keysH[((size_t)mt*16 + m)*256 + n] = (_Float16)(acc[r] + bk[n]);
  }
}

// ---------------- per-step kernel 1: prenet + attention GRU + query ----------------

__global__ void __launch_bounds__(256) k_step1(
    float* __restrict__ h_att, const float* __restrict__ ctx,
    const float* __restrict__ last_out, float* __restrict__ qbuf,
    const _Float16* pW1, const float* pb1, const _Float16* pW2, const float* pb2,
    const _Float16* aWg, const float* abg, const _Float16* aWc, const float* abc,
    const _Float16* pWq, const float* bq)
{
  __shared__ __align__(16) _Float16 sA [16*640];
  __shared__ __align__(16) _Float16 sP1[16*256];
  __shared__ __align__(16) _Float16 sP [16*128];
  __shared__ float sU[16*256];
  const int tid = threadIdx.x, lane = tid & 31, wave = tid >> 5;
  const int r0 = blockIdx.x * 16;

  // last_out -> 16x96 (zero padded)
  for (int i = tid; i < 16*96; i += 256){
    int m = i/96, k = i%96;
    sA[i] = (k < MEL_) ? (_Float16)last_out[(r0+m)*MEL_ + k] : (_Float16)0.f;
  }
  __syncthreads();
  // prenet1: relu(last @ W1 + b1) -> sP1 (16x256)
  for (int i = 0; i < 2; ++i){
    int nt = wave + i*8;
    v8f acc = wmma_tile(sA, 96, pW1, 3, nt, lane);
    #pragma unroll
    for (int r = 0; r < 8; ++r){
      int m = (lane<16)? r : r+8, n = nt*16 + (lane&15);
      float v = acc[r] + pb1[n]; v = v > 0.f ? v : 0.f;
      sP1[m*256+n] = (_Float16)v;
    }
  }
  __syncthreads();
  // prenet2: relu(p1 @ W2 + b2) -> sP (16x128)
  {
    int nt = wave;
    v8f acc = wmma_tile(sP1, 256, pW2, 8, nt, lane);
    #pragma unroll
    for (int r = 0; r < 8; ++r){
      int m = (lane<16)? r : r+8, n = nt*16 + (lane&15);
      float v = acc[r] + pb2[n]; v = v > 0.f ? v : 0.f;
      sP[m*128+n] = (_Float16)v;
    }
  }
  __syncthreads();
  // xg = [ctx | p | h_att] (16 x 640)
  for (int i = tid; i < 16*640; i += 256){
    int m = i/640, c = i%640;
    float v;
    if (c < 256)      v = ctx[(r0+m)*256 + c];
    else if (c < 384) v = (float)sP[m*128 + (c-256)];
    else              v = h_att[(r0+m)*256 + (c-384)];
    sA[i] = (_Float16)v;
  }
  __syncthreads();
  // gates (N=512), staged in regs, then overwrite h slot with r*h
  {
    v8f ga[4];
    for (int i = 0; i < 4; ++i) ga[i] = wmma_tile(sA, 640, aWg, 20, wave*4+i, lane);
    __syncthreads();
    for (int i = 0; i < 4; ++i){
      int nt = wave*4+i;
      #pragma unroll
      for (int r = 0; r < 8; ++r){
        int m = (lane<16)? r : r+8, n = nt*16 + (lane&15);
        float g = fast_sigmoid(ga[i][r] + abg[n]);
        if (n < 256) sA[m*640 + 384 + n] = (_Float16)(g * h_att[(r0+m)*256 + n]);
        else         sU[m*256 + (n-256)] = g;
      }
    }
  }
  __syncthreads();
  // candidate + h_att update (writes go to sP1/global, not to sA => safe)
  for (int i = 0; i < 2; ++i){
    int nt = wave*2+i;
    v8f acc = wmma_tile(sA, 640, aWc, 20, nt, lane);
    #pragma unroll
    for (int r = 0; r < 8; ++r){
      int m = (lane<16)? r : r+8, n = nt*16 + (lane&15);
      float c = fast_tanh(acc[r] + abc[n]);
      float u = sU[m*256+n];
      float hn = u*h_att[(r0+m)*256+n] + (1.f-u)*c;
      h_att[(r0+m)*256+n] = hn;
      sP1[m*256+n] = (_Float16)hn;
    }
  }
  __syncthreads();
  // q = h_att @ Wq + bq
  for (int i = 0; i < 2; ++i){
    int nt = wave*2+i;
    v8f acc = wmma_tile(sP1, 256, pWq, 8, nt, lane);
    #pragma unroll
    for (int r = 0; r < 8; ++r){
      int m = (lane<16)? r : r+8, n = nt*16 + (lane&15);
      qbuf[(r0+m)*256+n] = acc[r] + bq[n];
    }
  }
}

// ---------------- per-step kernel 2: energies + softmax + context (per batch row) ----

__global__ void __launch_bounds__(256) k_attn(
    const float* __restrict__ qbuf, const _Float16* __restrict__ keysH,
    const _Float16* __restrict__ encH, const float* __restrict__ ss,
    float* __restrict__ ctx, const int* __restrict__ inp_mask,
    const float* __restrict__ vvec, float* __restrict__ alphas_out,
    float* __restrict__ wp, int step)
{
  __shared__ float sQ[256], sV[256], sE[1024], red[256];
  const int tid = threadIdx.x, b = blockIdx.x;
  sQ[tid] = qbuf[b*256 + tid];
  sV[tid] = vvec[tid];
  __syncthreads();
  const int tm = inp_mask[b];
  for (int tt = 0; tt < 4; ++tt){
    int t = tid + tt*256;
    const _Float16* kr = keysH + (((size_t)b << 10) + t) * 256;
    float acc = 0.f;
    for (int d8 = 0; d8 < 32; ++d8){
      v8h kv = *(const v8h*)(kr + d8*8);
      #pragma unroll
      for (int j = 0; j < 8; ++j){
        int d = d8*8 + j;
        acc += fast_tanh((float)kv[j] + sQ[d]) * sV[d];
      }
    }
    sE[t] = (t < tm) ? acc : -1.0e9f;
  }
  __syncthreads();
  // max
  red[tid] = fmaxf(fmaxf(sE[tid], sE[tid+256]), fmaxf(sE[tid+512], sE[tid+768]));
  __syncthreads();
  for (int s = 128; s > 0; s >>= 1){ if (tid < s) red[tid] = fmaxf(red[tid], red[tid+s]); __syncthreads(); }
  float mx = red[0];
  __syncthreads();
  // exp + sum
  float psum = 0.f;
  for (int tt = 0; tt < 4; ++tt){
    int t = tid + tt*256;
    float a = __builtin_amdgcn_exp2f((sE[t] - mx) * 1.4426950408889634f);
    sE[t] = a; psum += a;
  }
  red[tid] = psum; __syncthreads();
  for (int s = 128; s > 0; s >>= 1){ if (tid < s) red[tid] += red[tid+s]; __syncthreads(); }
  float inv = 1.0f / red[0];
  __syncthreads();
  for (int tt = 0; tt < 4; ++tt){
    int t = tid + tt*256;
    float a = sE[t] * inv;
    sE[t] = a;
    alphas_out[(size_t)step*(B_*TENC) + (size_t)b*TENC + t] = a;
  }
  __syncthreads();
  // context: thread owns one d column (coalesced over L2-resident enc f16)
  const _Float16* er = encH + ((size_t)b << 10) * 256 + tid;
  float acc = 0.f;
  for (int t = 0; t < 1024; ++t) acc += sE[t] * (float)er[(size_t)t*256];
  ctx[b*256 + tid] = acc;
  // weight_per partial
  float as = fabsf(ss[b*256 + tid]);
  red[tid] = as / (fabsf(acc) + as);
  __syncthreads();
  for (int s = 128; s > 0; s >>= 1){ if (tid < s) red[tid] += red[tid+s]; __syncthreads(); }
  if (tid == 0) atomicAdd(&wp[step], red[0] * (1.0f/32768.0f));
}

// ---------------- per-step kernel 3: deci + GRU1 + GRU2 + output projection -------

__global__ void __launch_bounds__(256) k_step3(
    const float* __restrict__ h_att, float* __restrict__ h1, float* __restrict__ h2,
    const float* __restrict__ ctx, const float* __restrict__ ss,
    float* __restrict__ last_out,
    const _Float16* dWi, const float* dbi,
    const _Float16* g1W, const float* g1b, const _Float16* c1W, const float* c1b,
    const _Float16* g2W, const float* g2b, const _Float16* c2W, const float* c2b,
    const _Float16* oW,  const float* ob,
    float* __restrict__ out_seq, int step)
{
  __shared__ __align__(16) _Float16 sX[16*512];
  __shared__ __align__(16) _Float16 sO[16*256];
  __shared__ float sT1[16*256];
  __shared__ float sT2[16*256];
  __shared__ float sU [16*256];
  const int tid = threadIdx.x, lane = tid & 31, wave = tid >> 5;
  const int r0 = blockIdx.x * 16;

  // xh = [h_att | ctx + ss]
  for (int i = tid; i < 16*512; i += 256){
    int m = i/512, c = i%512;
    float v = (c < 256) ? h_att[(r0+m)*256 + c]
                        : (ctx[(r0+m)*256 + (c-256)] + ss[(r0+m)*256 + (c-256)]);
    sX[i] = (_Float16)v;
  }
  __syncthreads();
  // dec_in = xh @ deci_W + b ; rebuild sX = [dec_in | h1]
  {
    v8f acc[2];
    for (int i = 0; i < 2; ++i) acc[i] = wmma_tile(sX, 512, dWi, 16, wave*2+i, lane);
    __syncthreads();
    for (int i = 0; i < 2; ++i){
      int nt = wave*2+i;
      #pragma unroll
      for (int r = 0; r < 8; ++r){
        int m = (lane<16)? r : r+8, n = nt*16 + (lane&15);
        float v = acc[i][r] + dbi[n];
        sT1[m*256+n] = v;
        sX[m*512+n]  = (_Float16)v;
      }
    }
    for (int i = tid; i < 16*256; i += 256){
      int m = i/256, c = i%256;
      sX[m*512 + 256 + c] = (_Float16)h1[(r0+m)*256 + c];
    }
  }
  __syncthreads();
  // GRU1 gates
  {
    v8f ga[4];
    for (int i = 0; i < 4; ++i) ga[i] = wmma_tile(sX, 512, g1W, 16, wave*4+i, lane);
    __syncthreads();
    for (int i = 0; i < 4; ++i){
      int nt = wave*4+i;
      #pragma unroll
      for (int r = 0; r < 8; ++r){
        int m = (lane<16)? r : r+8, n = nt*16 + (lane&15);
        float g = fast_sigmoid(ga[i][r] + g1b[n]);
        if (n < 256) sX[m*512 + 256 + n] = (_Float16)(g * h1[(r0+m)*256 + n]);
        else         sU[m*256 + (n-256)] = g;
      }
    }
  }
  __syncthreads();
  // GRU1 candidate, h1 update, o1 = h1 + dec_in ; prep sX = [o1 | h2]
  {
    v8f ca[2];
    for (int i = 0; i < 2; ++i) ca[i] = wmma_tile(sX, 512, c1W, 16, wave*2+i, lane);
    __syncthreads();
    for (int i = 0; i < 2; ++i){
      int nt = wave*2+i;
      #pragma unroll
      for (int r = 0; r < 8; ++r){
        int m = (lane<16)? r : r+8, n = nt*16 + (lane&15);
        float c = fast_tanh(ca[i][r] + c1b[n]);
        float u = sU[m*256+n];
        float hn = u*h1[(r0+m)*256+n] + (1.f-u)*c;
        h1[(r0+m)*256+n] = hn;
        float o1 = hn + sT1[m*256+n];
        sT2[m*256+n] = o1;
        sX[m*512+n]  = (_Float16)o1;
      }
    }
    for (int i = tid; i < 16*256; i += 256){
      int m = i/256, c = i%256;
      sX[m*512 + 256 + c] = (_Float16)h2[(r0+m)*256 + c];
    }
  }
  __syncthreads();
  // GRU2 gates
  {
    v8f ga[4];
    for (int i = 0; i < 4; ++i) ga[i] = wmma_tile(sX, 512, g2W, 16, wave*4+i, lane);
    __syncthreads();
    for (int i = 0; i < 4; ++i){
      int nt = wave*4+i;
      #pragma unroll
      for (int r = 0; r < 8; ++r){
        int m = (lane<16)? r : r+8, n = nt*16 + (lane&15);
        float g = fast_sigmoid(ga[i][r] + g2b[n]);
        if (n < 256) sX[m*512 + 256 + n] = (_Float16)(g * h2[(r0+m)*256 + n]);
        else         sU[m*256 + (n-256)] = g;
      }
    }
  }
  __syncthreads();
  // GRU2 candidate, h2 update, o2 = h2 + o1 -> sO
  {
    v8f ca[2];
    for (int i = 0; i < 2; ++i) ca[i] = wmma_tile(sX, 512, c2W, 16, wave*2+i, lane);
    __syncthreads();
    for (int i = 0; i < 2; ++i){
      int nt = wave*2+i;
      #pragma unroll
      for (int r = 0; r < 8; ++r){
        int m = (lane<16)? r : r+8, n = nt*16 + (lane&15);
        float c = fast_tanh(ca[i][r] + c2b[n]);
        float u = sU[m*256+n];
        float hn = u*h2[(r0+m)*256+n] + (1.f-u)*c;
        h2[(r0+m)*256+n] = hn;
        sO[m*256+n] = (_Float16)(hn + sT2[m*256+n]);
      }
    }
  }
  __syncthreads();
  // dense = o2 @ out_W + out_b  (N=400 -> 25 tiles)
  for (int nt = wave; nt < 25; nt += 8){
    v8f acc = wmma_tile(sO, 256, oW, 8, nt, lane);
    #pragma unroll
    for (int r = 0; r < 8; ++r){
      int m = (lane<16)? r : r+8, n = nt*16 + (lane&15);
      float v = acc[r] + ob[n];
      out_seq[(size_t)(r0+m)*(TDEC*NOUT) + (size_t)step*NOUT + n] = v;
      if (n >= NOUT-MEL_) last_out[(r0+m)*MEL_ + (n - (NOUT-MEL_))] = v;
    }
  }
}

// ---------------- host ----------------

extern "C" void kernel_launch(void* const* d_in, const int* in_sizes, int n_in,
                              void* d_out, int out_size, void* d_ws, size_t ws_size,
                              hipStream_t stream)
{
  const float* enc      = (const float*)d_in[0];
  const int*   inp_mask = (const int*)  d_in[1];
  const float* inp_att  = (const float*)d_in[2];
  const float* style_tk = (const float*)d_in[3];
  const float* pre_W1   = (const float*)d_in[4];
  const float* pre_b1   = (const float*)d_in[5];
  const float* pre_W2   = (const float*)d_in[6];
  const float* pre_b2   = (const float*)d_in[7];
  const float* att_Wg   = (const float*)d_in[8];
  const float* att_bg   = (const float*)d_in[9];
  const float* att_Wc   = (const float*)d_in[10];
  const float* att_bc   = (const float*)d_in[11];
  const float* attn_Wk  = (const float*)d_in[12];
  const float* attn_bk  = (const float*)d_in[13];
  const float* attn_Wq  = (const float*)d_in[14];
  const float* attn_bq  = (const float*)d_in[15];
  const float* attn_v   = (const float*)d_in[16];
  const float* deci_W   = (const float*)d_in[17];
  const float* deci_b   = (const float*)d_in[18];
  const float* dec1_Wg  = (const float*)d_in[19];
  const float* dec1_bg  = (const float*)d_in[20];
  const float* dec1_Wc  = (const float*)d_in[21];
  const float* dec1_bc  = (const float*)d_in[22];
  const float* dec2_Wg  = (const float*)d_in[23];
  const float* dec2_bg  = (const float*)d_in[24];
  const float* dec2_Wc  = (const float*)d_in[25];
  const float* dec2_bc  = (const float*)d_in[26];
  const float* out_W    = (const float*)d_in[27];
  const float* out_b    = (const float*)d_in[28];

  char* ws = (char*)d_ws;
  size_t off = 0;
  auto alloc = [&](size_t bytes)->void*{
    size_t o = off; off = (off + bytes + 255) & ~(size_t)255; return (void*)(ws + o);
  };
  _Float16* encH  = (_Float16*)alloc((size_t)B_*TENC*DENC*2);   // 64 MB, L2-resident
  _Float16* keysH = (_Float16*)alloc((size_t)B_*TENC*DENC*2);   // 64 MB, L2-resident
  float* ssS  = (float*)alloc(B_*DENC*4);
  float* hAtt = (float*)alloc(B_*DENC*4);
  float* h1S  = (float*)alloc(B_*DENC*4);
  float* h2S  = (float*)alloc(B_*DENC*4);
  float* ctxS = (float*)alloc(B_*DENC*4);
  float* qS   = (float*)alloc(B_*DENC*4);
  float* loS  = (float*)alloc(B_*MEL_*4);
  auto packbuf = [&](int KT, int N){ return (_Float16*)alloc((size_t)KT*32*N*2); };
  _Float16* pW1 = packbuf(3, 256);
  _Float16* pW2 = packbuf(8, 128);
  _Float16* aWg = packbuf(20, 512);
  _Float16* aWc = packbuf(20, 256);
  _Float16* pWk = packbuf(8, 256);
  _Float16* pWq = packbuf(8, 256);
  _Float16* dWi = packbuf(16, 256);
  _Float16* g1W = packbuf(16, 512);
  _Float16* c1W = packbuf(16, 256);
  _Float16* g2W = packbuf(16, 512);
  _Float16* c2W = packbuf(16, 256);
  _Float16* oWp = packbuf(8, 400);

  float* outF    = (float*)d_out;
  float* alphasO = outF + (size_t)B_*TDEC*NOUT;       // 12,800,000
  float* wpO     = alphasO + (size_t)TDEC*B_*TENC;    // +32,768,000

  k_init<<<128, 256, 0, stream>>>(inp_att, style_tk, ssS, hAtt, h1S, h2S, ctxS, loS, wpO);
  {
    size_t n = (size_t)B_*TENC*DENC;
    k_cvt<<<(unsigned)((n + 255)/256), 256, 0, stream>>>(enc, encH, n);
  }
  auto pack = [&](const float* W, _Float16* o, int K, int N, int KT){
    int tot = KT*32*N;
    k_pack<<<(tot + 255)/256, 256, 0, stream>>>(W, o, K, N, KT);
  };
  pack(pre_W1,  pW1, 80,  256, 3);
  pack(pre_W2,  pW2, 256, 128, 8);
  pack(att_Wg,  aWg, 640, 512, 20);
  pack(att_Wc,  aWc, 640, 256, 20);
  pack(attn_Wk, pWk, 256, 256, 8);
  pack(attn_Wq, pWq, 256, 256, 8);
  pack(deci_W,  dWi, 512, 256, 16);
  pack(dec1_Wg, g1W, 512, 512, 16);
  pack(dec1_Wc, c1W, 512, 256, 16);
  pack(dec2_Wg, g2W, 512, 512, 16);
  pack(dec2_Wc, c2W, 512, 256, 16);
  pack(out_W,   oWp, 256, 400, 8);

  k_keys<<<16384, 256, 0, stream>>>(encH, pWk, attn_bk, keysH);

  for (int step = 0; step < TDEC; ++step){
    k_step1<<<8, 256, 0, stream>>>(hAtt, ctxS, loS, qS, pW1, pre_b1, pW2, pre_b2,
                                   aWg, att_bg, aWc, att_bc, pWq, attn_bq);
    k_attn<<<128, 256, 0, stream>>>(qS, keysH, encH, ssS, ctxS, inp_mask, attn_v,
                                    alphasO, wpO, step);
    k_step3<<<8, 256, 0, stream>>>(hAtt, h1S, h2S, ctxS, ssS, loS,
                                   dWi, deci_b, g1W, dec1_bg, c1W, dec1_bc,
                                   g2W, dec2_bg, c2W, dec2_bc, oWp, out_b,
                                   outF, step);
  }
  (void)in_sizes; (void)n_in; (void)out_size; (void)ws_size;
}
